// LSTMController_77610059038881
// MI455X (gfx1250) — matrix operand order repaired and checked by reading, not compiled, not run on hardware
//
#include <hip/hip_runtime.h>
#include <hip/hip_bf16.h>

#define B_      128
#define H_      512
#define T_ENC_  256
#define T_OUT_  180
#define G4H     2048          // 4*H gate columns
#define NWG     128           // persistent workgroups (each owns 4 hidden units)
#define THREADS 256           // 8 wave32
#define BH      (B_ * H_)     // elements in one h-state buffer
#define GSTRIDE 17            // LDS gate-stage row stride (floats), conflict-free

typedef unsigned short ushort_t;
typedef __attribute__((ext_vector_type(16))) __bf16 v16bf;
typedef __attribute__((ext_vector_type(8)))  float  v8f;

union V16U { uint4 q[2]; v16bf v; };

__device__ __forceinline__ ushort_t f2bf(float f) {
    unsigned u = __float_as_uint(f);
    u += 0x7FFFu + ((u >> 16) & 1u);          // round-to-nearest-even
    return (ushort_t)(u >> 16);
}
__device__ __forceinline__ float bf2f(ushort_t h) {
    return __uint_as_float(((unsigned)h) << 16);
}
__device__ __forceinline__ float sigm(float x) { return 1.0f / (1.0f + __expf(-x)); }

// ---- A operand: 16x32 bf16 tile from row-major h[128][512] (bf16) ----------
// CDNA5 layout: lane<16 -> row=lane,   K chunks [0..7] and [16..23]
//               lane>=16 -> row=lane-16, K chunks [8..15] and [24..31]
__device__ __forceinline__ v16bf load_a(const ushort_t* __restrict__ src,
                                        int row, int k0, int lane) {
    int c = (lane & 16) ? 8 : 0;
    const ushort_t* p = src + (size_t)row * H_ + k0 + c;
    V16U u;
    u.q[0] = *(const uint4*)(p);
    u.q[1] = *(const uint4*)(p + 16);
    return u.v;
}

// ---- B operand: 32x16 bf16 tile from LDS weights stored [col][K(+8 pad)] ----
// lane<16: col=lane, K chunks [0..7],[16..23]; lane>=16: col=lane-16, K chunks [8..15],[24..31]
__device__ __forceinline__ v16bf load_b(const ushort_t* __restrict__ wl,
                                        int stride, int k0, int lane) {
    const ushort_t* p = wl + (lane & 15) * stride + k0 + ((lane & 16) ? 8 : 0);
    V16U u;
    u.q[0] = *(const uint4*)(p);
    u.q[1] = *(const uint4*)(p + 16);
    return u.v;
}

__device__ __forceinline__ v8f tile_gemm(const ushort_t* __restrict__ A0, int n0,
                                         const ushort_t* __restrict__ A1, int n1,
                                         const ushort_t* __restrict__ Bl, int bstride,
                                         int rowA, int lane) {
    v8f acc = {0.f, 0.f, 0.f, 0.f, 0.f, 0.f, 0.f, 0.f};
    int kb = 0;
    for (int i = 0; i < n0; ++i, kb += 32) {
        v16bf a = load_a(A0, rowA, 32 * i, lane);
        v16bf b = load_b(Bl, bstride, kb, lane);
        acc = __builtin_amdgcn_wmma_f32_16x16x32_bf16(false, a, false, b,
                                                      (short)0, acc, false, false);
    }
    for (int i = 0; i < n1; ++i, kb += 32) {
        v16bf a = load_a(A1, rowA, 32 * i, lane);
        v16bf b = load_b(Bl, bstride, kb, lane);
        acc = __builtin_amdgcn_wmma_f32_16x16x32_bf16(false, a, false, b,
                                                      (short)0, acc, false, false);
    }
    return acc;
}

// ---- grid-wide barrier (persistent kernel), device-scope atomics ------------
__device__ __forceinline__ void grid_barrier(unsigned* cnt, unsigned* gen, unsigned* lgen) {
    __threadfence();
    __syncthreads();
    if (threadIdx.x == 0) {
        unsigned g = ++(*lgen);
        unsigned old = __hip_atomic_fetch_add(cnt, 1u, __ATOMIC_ACQ_REL,
                                              __HIP_MEMORY_SCOPE_AGENT);
        if (old == NWG - 1) {
            __hip_atomic_store(cnt, 0u, __ATOMIC_RELAXED, __HIP_MEMORY_SCOPE_AGENT);
            __hip_atomic_store(gen, g, __ATOMIC_RELEASE, __HIP_MEMORY_SCOPE_AGENT);
        } else {
            while (__hip_atomic_load(gen, __ATOMIC_ACQUIRE,
                                     __HIP_MEMORY_SCOPE_AGENT) < g)
                __builtin_amdgcn_s_sleep(2);
        }
    }
    __syncthreads();
}

// ---- init: zero barrier counters + h ping-pong buffers ----------------------
__global__ void init_zero(unsigned* p, int n) {
    int i = blockIdx.x * blockDim.x + threadIdx.x;
    if (i < n) p[i] = 0u;
}

// ---- weight repack: fp32 -> bf16, concat K, fold decoder rank-1 feedback ----
__global__ void repack(const float* __restrict__ eWhh0, const float* __restrict__ eWih1,
                       const float* __restrict__ eWhh1, const float* __restrict__ dWih0,
                       const float* __restrict__ dWhh0, const float* __restrict__ dWih1,
                       const float* __restrict__ dWhh1,
                       const float* __restrict__ eBih0, const float* __restrict__ eBhh0,
                       const float* __restrict__ eBih1, const float* __restrict__ eBhh1,
                       const float* __restrict__ dBih0, const float* __restrict__ dBhh0,
                       const float* __restrict__ dBih1, const float* __restrict__ dBhh1,
                       const float* __restrict__ outW, const float* __restrict__ outb,
                       ushort_t* __restrict__ WE0, ushort_t* __restrict__ WE1,
                       ushort_t* __restrict__ WD0, ushort_t* __restrict__ WD1,
                       float* __restrict__ biasg) {
    int c = blockIdx.x;              // gate column 0..2047
    int t = threadIdx.x;
    for (int k = t; k < 512; k += THREADS)
        WE0[(size_t)c * 512 + k] = f2bf(eWhh0[(size_t)c * 512 + k]);
    for (int k = t; k < 1024; k += THREADS) {
        // enc L1: A = [h0 | h1]  ->  [Wih1 | Whh1]
        WE1[(size_t)c * 1024 + k] =
            f2bf((k < 512) ? eWih1[(size_t)c * 512 + k] : eWhh1[(size_t)c * 512 + k - 512]);
        // dec L0: A = [h0 | h1]  ->  [Whh0 | outW^T (x) Wih0]  (rank-1 feedback fold)
        WD0[(size_t)c * 1024 + k] =
            f2bf((k < 512) ? dWhh0[(size_t)c * 512 + k] : outW[k - 512] * dWih0[c]);
        // dec L1: A = [h0 | h1]  ->  [Wih1 | Whh1]
        WD1[(size_t)c * 1024 + k] =
            f2bf((k < 512) ? dWih1[(size_t)c * 512 + k] : dWhh1[(size_t)c * 512 + k - 512]);
    }
    if (t == 0) {
        biasg[c]            = eBih0[c] + eBhh0[c];
        biasg[2048 + c]     = eBih1[c] + eBhh1[c];
        biasg[4096 + c]     = dBih0[c] + dBhh0[c];
        biasg[6144 + c]     = outb[0] * dWih0[c];      // extra bias when y_prev active
        biasg[8192 + c]     = dBih1[c] + dBhh1[c];
    }
}

// ---- persistent LSTM: weight-stationary in LDS, c in registers --------------
__global__ __launch_bounds__(THREADS, 1)
void lstm_persistent(const float* __restrict__ x,
                     const float* __restrict__ encWih0,      // [2048][8] fp32
                     const ushort_t* __restrict__ WE0g,      // [2048][512]  bf16
                     const ushort_t* __restrict__ WE1g,      // [2048][1024] bf16
                     const ushort_t* __restrict__ WD0g,      // [2048][1024] bf16
                     const ushort_t* __restrict__ WD1g,      // [2048][1024] bf16
                     const float* __restrict__ biasg,        // 5 x 2048 fp32
                     ushort_t* __restrict__ h0,              // 2 x BH bf16 ping-pong
                     ushort_t* __restrict__ h1e,             // 2 x BH bf16 ping-pong
                     ushort_t* __restrict__ hist,            // (T_OUT+1) x BH bf16
                     unsigned* bar_cnt, unsigned* bar_gen) {
    extern __shared__ char smem[];
    ushort_t* wE0   = (ushort_t*)smem;                 // 16 x (512+8)
    ushort_t* wE1   = wE0 + 16 * 520;                  // 16 x (1024+8)
    ushort_t* wD0   = wE1 + 16 * 1032;
    ushort_t* wD1   = wD0 + 16 * 1032;
    float*    gst   = (float*)(wD1 + 16 * 1032);       // 128 x 17 gate staging
    float*    wih0_l = gst + 128 * GSTRIDE;            // 16 x 8 fp32
    float*    bE0_l = wih0_l + 128;
    float*    bE1_l = bE0_l + 16;
    float*    bD0_l = bE1_l + 16;
    float*    bD0x_l = bD0_l + 16;
    float*    bD1_l = bD0x_l + 16;

    const int lane = threadIdx.x & 31;
    const int wave = threadIdx.x >> 5;
    const int rowA = 16 * wave + (lane & 15);
    const int wg = blockIdx.x;
    const int colbase = wg * 4;                        // this WG's 4 hidden units

    // ---- load weight slices into LDS (once) ----
    auto load_slice = [&](ushort_t* dst, int stride, const ushort_t* src, int K) {
        for (int idx = threadIdx.x; idx < 16 * K; idx += THREADS) {
            int lc = idx / K, k = idx - lc * K;
            int gc = (lc >> 2) * 512 + colbase + (lc & 3);
            dst[lc * stride + k] = src[(size_t)gc * K + k];
        }
    };
    load_slice(wE0, 520, WE0g, 512);
    load_slice(wE1, 1032, WE1g, 1024);
    load_slice(wD0, 1032, WD0g, 1024);
    load_slice(wD1, 1032, WD1g, 1024);
    for (int idx = threadIdx.x; idx < 16 * 8; idx += THREADS) {
        int lc = idx >> 3, k = idx & 7;
        int gc = (lc >> 2) * 512 + colbase + (lc & 3);
        wih0_l[idx] = encWih0[gc * 8 + k];
    }
    for (int lc = threadIdx.x; lc < 16; lc += THREADS) {
        int gc = (lc >> 2) * 512 + colbase + (lc & 3);
        bE0_l[lc]  = biasg[gc];
        bE1_l[lc]  = biasg[2048 + gc];
        bD0_l[lc]  = biasg[4096 + gc];
        bD0x_l[lc] = biasg[6144 + gc];
        bD1_l[lc]  = biasg[8192 + gc];
    }
    __syncthreads();

    float c0v[2] = {0.f, 0.f};      // layer-0 cell state (2 (b,u) pairs / thread)
    float c1v[2] = {0.f, 0.f};      // layer-1 cell state
    unsigned lgen = 0;

    auto run_cell = [&](const ushort_t* A0, int n0, const ushort_t* A1, int n1,
                        const ushort_t* Wl, int ws, const float* bias_l,
                        const float* bias2_l, const float* xbase,
                        float* cv, ushort_t* hdst) {
        v8f acc = tile_gemm(A0, n0, A1, n1, Wl, ws, rowA, lane);
        {   // stage C tile: VGPR i = row (i + 8*(lane>=16)), N = lane%16
            int col = lane & 15;
            int rb = 16 * wave + ((lane & 16) ? 8 : 0);
#pragma unroll
            for (int i = 0; i < 8; ++i) gst[(rb + i) * GSTRIDE + col] = acc[i];
        }
        __syncthreads();
#pragma unroll
        for (int r = 0; r < 2; ++r) {
            int p = threadIdx.x + THREADS * r;
            int b = p & (B_ - 1);
            int u = p >> 7;                                // 0..3
            float gi = gst[b * GSTRIDE + u]       + bias_l[u];
            float gf = gst[b * GSTRIDE + 4 + u]   + bias_l[4 + u];
            float gg = gst[b * GSTRIDE + 8 + u]   + bias_l[8 + u];
            float go = gst[b * GSTRIDE + 12 + u]  + bias_l[12 + u];
            if (bias2_l) {
                gi += bias2_l[u]; gf += bias2_l[4 + u];
                gg += bias2_l[8 + u]; go += bias2_l[12 + u];
            }
            if (xbase) {                                   // encoder L0: x@Wih0^T
                const float* xp = xbase + (size_t)b * (T_ENC_ * 8);
#pragma unroll
                for (int k = 0; k < 8; ++k) {
                    float xv = xp[k];
                    gi += xv * wih0_l[u * 8 + k];
                    gf += xv * wih0_l[(4 + u) * 8 + k];
                    gg += xv * wih0_l[(8 + u) * 8 + k];
                    go += xv * wih0_l[(12 + u) * 8 + k];
                }
            }
            float c2 = sigm(gf) * cv[r] + sigm(gi) * tanhf(gg);
            float h2 = sigm(go) * tanhf(c2);
            cv[r] = c2;
            hdst[(size_t)b * H_ + colbase + u] = f2bf(h2);
        }
    };

    // ================= encoder =================
    for (int t = 0; t < T_ENC_; ++t) {
        const ushort_t* h0p = h0 + (size_t)(t & 1) * BH;
        ushort_t*       h0c = h0 + (size_t)((t + 1) & 1) * BH;
        const ushort_t* h1p = h1e + (size_t)(t & 1) * BH;
        ushort_t*       h1c = h1e + (size_t)((t + 1) & 1) * BH;
        run_cell(h0p, 16, (const ushort_t*)0, 0, wE0, 520, bE0_l,
                 (const float*)0, x + t * 8, c0v, h0c);
        grid_barrier(bar_cnt, bar_gen, &lgen);
        run_cell(h0c, 16, h1p, 16, wE1, 1032, bE1_l,
                 (const float*)0, (const float*)0, c1v, h1c);
        grid_barrier(bar_cnt, bar_gen, &lgen);
    }
    // ================= decoder =================
    // encoder final: h0 in slot (T_ENC&1)=0, h1 in h1e slot 0; c carried in regs
    for (int s = 0; s < T_OUT_; ++s) {
        int t = T_ENC_ + s;
        const ushort_t* h0p = h0 + (size_t)(t & 1) * BH;
        ushort_t*       h0c = h0 + (size_t)((t + 1) & 1) * BH;
        const ushort_t* h1p = (s == 0) ? h1e : (hist + (size_t)s * BH);
        ushort_t*       h1c = hist + (size_t)(s + 1) * BH;
        // cell0: y_prev folded as rank-1 over h1_prev; step 0 has y_prev = 0
        run_cell(h0p, 16, h1p, (s == 0) ? 0 : 16, wD0, 1032, bD0_l,
                 (s == 0) ? (const float*)0 : bD0x_l, (const float*)0, c0v, h0c);
        grid_barrier(bar_cnt, bar_gen, &lgen);
        run_cell(h0c, 16, h1p, 16, wD1, 1032, bD1_l,
                 (const float*)0, (const float*)0, c1v, h1c);
        grid_barrier(bar_cnt, bar_gen, &lgen);
    }
}

// ---- final projection: y[b,s] = hist[s+1][b,:] . outW + out_b ---------------
__global__ void out_proj(const ushort_t* __restrict__ hist, const float* __restrict__ outW,
                         const float* __restrict__ outb, float* __restrict__ y) {
    int idx = blockIdx.x * blockDim.x + threadIdx.x;
    if (idx >= B_ * T_OUT_) return;
    int b = idx / T_OUT_;
    int s = idx - b * T_OUT_;
    const ushort_t* h = hist + (size_t)(s + 1) * BH + (size_t)b * H_;
    float acc = 0.f;
    for (int k = 0; k < H_; ++k) acc += bf2f(h[k]) * outW[k];
    y[idx] = acc + outb[0];
}

extern "C" void kernel_launch(void* const* d_in, const int* in_sizes, int n_in,
                              void* d_out, int out_size, void* d_ws, size_t ws_size,
                              hipStream_t stream) {
    (void)in_sizes; (void)n_in; (void)out_size; (void)ws_size;
    const float* x     = (const float*)d_in[0];
    const float* eWih0 = (const float*)d_in[1];
    const float* eWhh0 = (const float*)d_in[2];
    const float* eBih0 = (const float*)d_in[3];
    const float* eBhh0 = (const float*)d_in[4];
    const float* eWih1 = (const float*)d_in[5];
    const float* eWhh1 = (const float*)d_in[6];
    const float* eBih1 = (const float*)d_in[7];
    const float* eBhh1 = (const float*)d_in[8];
    const float* dWih0 = (const float*)d_in[9];
    const float* dWhh0 = (const float*)d_in[10];
    const float* dBih0 = (const float*)d_in[11];
    const float* dBhh0 = (const float*)d_in[12];
    const float* dWih1 = (const float*)d_in[13];
    const float* dWhh1 = (const float*)d_in[14];
    const float* dBih1 = (const float*)d_in[15];
    const float* dBhh1 = (const float*)d_in[16];
    const float* outW  = (const float*)d_in[17];
    const float* outb  = (const float*)d_in[18];
    float* y = (float*)d_out;

    char* ws = (char*)d_ws;
    size_t off = 0;
    auto carve = [&](size_t bytes) -> char* {
        char* p = ws + off;
        off = (off + bytes + 255) & ~(size_t)255;
        return p;
    };
    unsigned* bar  = (unsigned*)carve(8);                          // cnt, gen
    ushort_t* h0   = (ushort_t*)carve((size_t)2 * BH * 2);
    ushort_t* h1e  = (ushort_t*)carve((size_t)2 * BH * 2);
    ushort_t* hist = (ushort_t*)carve((size_t)(T_OUT_ + 1) * BH * 2);
    ushort_t* WE0g = (ushort_t*)carve((size_t)G4H * 512 * 2);
    ushort_t* WE1g = (ushort_t*)carve((size_t)G4H * 1024 * 2);
    ushort_t* WD0g = (ushort_t*)carve((size_t)G4H * 1024 * 2);
    ushort_t* WD1g = (ushort_t*)carve((size_t)G4H * 1024 * 2);
    float*    biasg = (float*)carve((size_t)5 * G4H * 4);

    // zero barrier counters + both h ping-pong regions (contiguous at ws start)
    const int zero_words = (256 + 2 * (2 * BH * 2)) / 4;           // 131136 u32
    init_zero<<<(zero_words + 255) / 256, 256, 0, stream>>>((unsigned*)ws, zero_words);

    repack<<<G4H, THREADS, 0, stream>>>(eWhh0, eWih1, eWhh1, dWih0, dWhh0, dWih1, dWhh1,
                                        eBih0, eBhh0, eBih1, eBhh1,
                                        dBih0, dBhh0, dBih1, dBhh1,
                                        outW, outb, WE0g, WE1g, WD0g, WD1g, biasg);

    const size_t lds = (size_t)(16 * 520 + 3 * 16 * 1032) * 2 +
                       (size_t)(128 * GSTRIDE + 16 * 8 + 5 * 16) * 4;   // 125,248 B
    lstm_persistent<<<NWG, THREADS, lds, stream>>>(x, eWih0, WE0g, WE1g, WD0g, WD1g,
                                                   biasg, h0, h1e, hist,
                                                   bar, bar + 1);

    out_proj<<<(B_ * T_OUT_ + 255) / 256, 256, 0, stream>>>(hist, outW, outb, y);
}
